// EquivariantDecoder_63608465654304
// MI455X (gfx1250) — matrix operand order repaired and checked by reading, not compiled, not run on hardware
//
#include <hip/hip_runtime.h>

// Problem constants (from the reference)
constexpr int HDIM = 256;
constexpr int NND  = 50000;
constexpr int NED  = 800000;
constexpr int W1PITCH = 258;  // bf16 elements per LDS row (pitch avoids bank conflicts)

typedef __attribute__((ext_vector_type(16))) __bf16 v16bf;
typedef __attribute__((ext_vector_type(8)))  float  v8f;

union BFrag { v16bf v; unsigned u[8]; };

__device__ __forceinline__ float silu_f(float v) {
  return v / (1.0f + __expf(-v));
}

// Stage W1 (f32 [K][N] row-major in global) into LDS transposed as bf16 [N][K] with pitch.
__device__ __forceinline__ void stage_w1(const float* __restrict__ W1, __bf16* sW1T) {
  for (int idx = threadIdx.x * 4; idx < HDIM * HDIM; idx += blockDim.x * 4) {
    const float4 v = *(const float4*)(W1 + idx);
    const int k = idx >> 8;    // K (row of W1)
    const int n = idx & 255;   // N (col of W1)
    sW1T[(n + 0) * W1PITCH + k] = (__bf16)v.x;
    sW1T[(n + 1) * W1PITCH + k] = (__bf16)v.y;
    sW1T[(n + 2) * W1PITCH + k] = (__bf16)v.z;
    sW1T[(n + 3) * W1PITCH + k] = (__bf16)v.w;
  }
}

// Load 8 A-fragments (16x32 bf16 each) covering K=0..255 for rows [arow tile].
// ISA layout (16-bit A 16x32): lanes 0-15 -> M=lane, K={0..7,16..23};
// lanes 16-31 -> M=lane-16, K={8..15,24..31}. VGPR i holds packed (K, K+1).
__device__ __forceinline__ void load_a_frags(const float* __restrict__ M, int arow,
                                             int khalf8, v16bf af[8]) {
  const float* rowp = M + (size_t)arow * HDIM;
#pragma unroll
  for (int kk = 0; kk < 8; ++kk) {
    v16bf a;
#pragma unroll
    for (int i = 0; i < 8; ++i) {
      const int k = kk * 32 + ((i < 4) ? (2 * i) : (8 + 2 * i)) + khalf8;
      const float2 v = *(const float2*)(rowp + k);
      a[2 * i]     = (__bf16)v.x;
      a[2 * i + 1] = (__bf16)v.y;
    }
    af[kk] = a;
  }
}

// Load a B fragment (32x16 bf16) from LDS-transposed W1.
// ISA layout (16-bit B 32x16): lanes 0-15 -> N=lane, K={0..7,16..23}; lanes 16-31 -> N=lane-16, K={8..15,24..31}.
// VGPR i holds packed (K, K+1) -> a single 32-bit LDS load per VGPR.
__device__ __forceinline__ v16bf load_b_frag(const __bf16* sW1T, int col, int kb, int kk) {
  BFrag f;
  const __bf16* base = sW1T + col * W1PITCH + kk * 32 + kb / 2;
#pragma unroll
  for (int i = 0; i < 8; ++i) {
    f.u[i] = *(const unsigned*)(base + ((i < 4) ? (2 * i) : (8 + 2 * i)));
  }
  return f.v;
}

// Reduce 8 per-row partials over the 16 column lanes, then distribute the 16 row
// scalars (plus bias) to lanes 0..15. Rows 0..7 end in lane 0, rows 8..15 in lane 16.
__device__ __forceinline__ float reduce_distribute(float part[8], int lane, float bias) {
#pragma unroll
  for (int r = 0; r < 8; ++r) {
    part[r] += __shfl_xor(part[r], 1, 32);
    part[r] += __shfl_xor(part[r], 2, 32);
    part[r] += __shfl_xor(part[r], 4, 32);
    part[r] += __shfl_xor(part[r], 8, 32);
  }
  float w = bias;
#pragma unroll
  for (int r = 0; r < 8; ++r) {
    const float wlo = __shfl(part[r], 0, 32);
    const float whi = __shfl(part[r], 16, 32);
    if (lane == r)     w += wlo;
    if (lane == r + 8) w += whi;
  }
  return w;
}

__device__ __forceinline__ void scatter_edges(const float* __restrict__ x,
                                              const int* __restrict__ eidx,
                                              float* __restrict__ sums,
                                              float* __restrict__ counts,
                                              int e0, int lane, float w) {
  if (lane < 16) {
    const int e = e0 + lane;
    const int s = eidx[e];
    const int d = eidx[NED + e];
    atomicAdd(&sums[3 * d + 0], (x[3 * s + 0] - x[3 * d + 0]) * w);
    atomicAdd(&sums[3 * d + 1], (x[3 * s + 1] - x[3 * d + 1]) * w);
    atomicAdd(&sums[3 * d + 2], (x[3 * s + 2] - x[3 * d + 2]) * w);
    atomicAdd(&counts[d], 1.0f);
  }
}

// ---------------- Kernel 1: edge-weight MLP + scatter of rel*w ----------------
// Each wave processes TWO 16-edge row tiles so every B fragment loaded from LDS
// feeds two independent WMMAs (halves LDS traffic per WMMA, doubles XDL ILP).
__global__ __launch_bounds__(128) void edge_weight_scatter(
    const float* __restrict__ mij, const float* __restrict__ x,
    const int* __restrict__ eidx,
    const float* __restrict__ W1, const float* __restrict__ b1,
    const float* __restrict__ W2, const float* __restrict__ b2,
    float* __restrict__ sums, float* __restrict__ counts) {
  extern __shared__ char smem[];
  __bf16* sW1T = (__bf16*)smem;
  float*  sb1  = (float*)(smem + (size_t)W1PITCH * HDIM * 2);
  float*  sw2  = sb1 + HDIM;

  stage_w1(W1, sW1T);
  for (int i = threadIdx.x; i < HDIM; i += blockDim.x) {
    sb1[i] = b1[i];
    sw2[i] = W2[i];
  }
  __syncthreads();

  const int lane   = threadIdx.x & 31;
  const int waveId = threadIdx.x >> 5;
  const int gw     = blockIdx.x * (blockDim.x >> 5) + waveId;
  const int nWaves = gridDim.x * (blockDim.x >> 5);
  const int col16  = lane & 15;
  const int khalf  = lane >> 4;
  const int kbB    = khalf * 16;  // B-fragment K base offset
  const int khalf8 = khalf * 8;   // A-fragment K base offset
  const float bias2 = b2[0];

  const int numPairs = NED / 32;  // 32 edges per wave iteration
  for (int pair = gw; pair < numPairs; pair += nWaves) {
    const int e0 = pair * 32;

    v16bf af0[8], af1[8];
    load_a_frags(mij, e0 + col16,      khalf8, af0);
    load_a_frags(mij, e0 + 16 + col16, khalf8, af1);

    float part0[8], part1[8];
#pragma unroll
    for (int r = 0; r < 8; ++r) { part0[r] = 0.0f; part1[r] = 0.0f; }

    for (int nt = 0; nt < 16; ++nt) {
      const int n0 = nt * 16;
      v8f acc0 = {0.f, 0.f, 0.f, 0.f, 0.f, 0.f, 0.f, 0.f};
      v8f acc1 = {0.f, 0.f, 0.f, 0.f, 0.f, 0.f, 0.f, 0.f};
#pragma unroll
      for (int kk = 0; kk < 8; ++kk) {
        const v16bf bf = load_b_frag(sW1T, n0 + col16, kbB, kk);
        acc0 = __builtin_amdgcn_wmma_f32_16x16x32_bf16(
            false, af0[kk], false, bf, (short)0, acc0, false, false);
        acc1 = __builtin_amdgcn_wmma_f32_16x16x32_bf16(
            false, af1[kk], false, bf, (short)0, acc1, false, false);
      }
      const int n = n0 + col16;
      const float bb  = sb1[n];
      const float w2n = sw2[n];
#pragma unroll
      for (int r = 0; r < 8; ++r) {
        part0[r] += silu_f(acc0[r] + bb) * w2n;
        part1[r] += silu_f(acc1[r] + bb) * w2n;
      }
    }

    const float w0 = reduce_distribute(part0, lane, bias2);
    const float w1 = reduce_distribute(part1, lane, bias2);
    scatter_edges(x, eidx, sums, counts, e0,      lane, w0);
    scatter_edges(x, eidx, sums, counts, e0 + 16, lane, w1);
  }
}

// ---------------- Kernel 2: node gate MLP + temporal velocity combine ----------------
__global__ __launch_bounds__(128) void node_gate_combine(
    const float* __restrict__ hin, const float* __restrict__ vel,
    const float* __restrict__ W1, const float* __restrict__ b1,
    const float* __restrict__ W2, const float* __restrict__ b2,
    float* __restrict__ out) {
  extern __shared__ char smem[];
  __bf16* sW1T = (__bf16*)smem;
  float*  sb1  = (float*)(smem + (size_t)W1PITCH * HDIM * 2);
  float*  sw2  = sb1 + HDIM;  // HDIM*5 floats

  stage_w1(W1, sW1T);
  for (int i = threadIdx.x; i < HDIM; i += blockDim.x) sb1[i] = b1[i];
  for (int i = threadIdx.x; i < HDIM * 5; i += blockDim.x) sw2[i] = W2[i];
  __syncthreads();

  const int lane   = threadIdx.x & 31;
  const int waveId = threadIdx.x >> 5;
  const int gw     = blockIdx.x * (blockDim.x >> 5) + waveId;
  const int nWaves = gridDim.x * (blockDim.x >> 5);
  const int col16  = lane & 15;
  const int khalf  = lane >> 4;
  const int kbB    = khalf * 16;
  const int khalf8 = khalf * 8;

  float b2r[5];
#pragma unroll
  for (int j = 0; j < 5; ++j) b2r[j] = b2[j];

  const int numTiles = NND / 16;
  for (int tile = gw; tile < numTiles; tile += nWaves) {
    const int n0row = tile * 16;

    v16bf af[8];
    load_a_frags(hin, n0row + col16, khalf8, af);

    float part[8][5];
#pragma unroll
    for (int r = 0; r < 8; ++r)
#pragma unroll
      for (int j = 0; j < 5; ++j) part[r][j] = 0.0f;

    for (int nt = 0; nt < 16; ++nt) {
      const int n0 = nt * 16;
      v8f acc = {0.f, 0.f, 0.f, 0.f, 0.f, 0.f, 0.f, 0.f};
#pragma unroll
      for (int kk = 0; kk < 8; ++kk) {
        const v16bf bf = load_b_frag(sW1T, n0 + col16, kbB, kk);
        acc = __builtin_amdgcn_wmma_f32_16x16x32_bf16(
            false, af[kk], false, bf, (short)0, acc, false, false);
      }
      const int n = n0 + col16;
      const float bb = sb1[n];
      float w2j[5];
#pragma unroll
      for (int j = 0; j < 5; ++j) w2j[j] = sw2[n * 5 + j];
#pragma unroll
      for (int r = 0; r < 8; ++r) {
        const float s = silu_f(acc[r] + bb);
#pragma unroll
        for (int j = 0; j < 5; ++j) part[r][j] += s * w2j[j];
      }
    }

#pragma unroll
    for (int r = 0; r < 8; ++r)
#pragma unroll
      for (int j = 0; j < 5; ++j) {
        part[r][j] += __shfl_xor(part[r][j], 1, 32);
        part[r][j] += __shfl_xor(part[r][j], 2, 32);
        part[r][j] += __shfl_xor(part[r][j], 4, 32);
        part[r][j] += __shfl_xor(part[r][j], 8, 32);
      }

    float aj[5];
#pragma unroll
    for (int j = 0; j < 5; ++j) aj[j] = b2r[j];
#pragma unroll
    for (int r = 0; r < 8; ++r)
#pragma unroll
      for (int j = 0; j < 5; ++j) {
        const float alo = __shfl(part[r][j], 0, 32);
        const float ahi = __shfl(part[r][j], 16, 32);
        if (lane == r)     aj[j] += alo;
        if (lane == r + 8) aj[j] += ahi;
      }

    if (lane < 16) {
      const int nd = n0row + lane;
      const float* vp = vel + (size_t)nd * 15;
      float o0 = 0.f, o1 = 0.f, o2 = 0.f;
#pragma unroll
      for (int j = 0; j < 5; ++j) {
        o0 += aj[j] * vp[j * 3 + 0];
        o1 += aj[j] * vp[j * 3 + 1];
        o2 += aj[j] * vp[j * 3 + 2];
      }
      out[3 * nd + 0] = o0;
      out[3 * nd + 1] = o1;
      out[3 * nd + 2] = o2;
    }
  }
}

// ---------------- Kernel 3: add scatter-mean geometry delta ----------------
__global__ __launch_bounds__(256) void finalize_add(
    float* __restrict__ out, const float* __restrict__ sums,
    const float* __restrict__ counts) {
  const int i = blockIdx.x * blockDim.x + threadIdx.x;
  if (i < 3 * NND) {
    const int n = i / 3;
    out[i] += sums[i] / fmaxf(counts[n], 1.0f);
  }
}

extern "C" void kernel_launch(void* const* d_in, const int* in_sizes, int n_in,
                              void* d_out, int out_size, void* d_ws, size_t ws_size,
                              hipStream_t stream) {
  const float* h_in   = (const float*)d_in[0];
  const float* m_ij   = (const float*)d_in[1];
  const float* x      = (const float*)d_in[2];
  const float* vel    = (const float*)d_in[3];
  const int*   eidx   = (const int*)d_in[4];   // JAX default: int32 (x64 disabled)
  const float* ew_W1  = (const float*)d_in[5];
  const float* ew_b1  = (const float*)d_in[6];
  const float* ew_W2  = (const float*)d_in[7];
  const float* ew_b2  = (const float*)d_in[8];
  const float* vg_W1  = (const float*)d_in[9];
  const float* vg_b1  = (const float*)d_in[10];
  const float* vg_W2  = (const float*)d_in[11];
  const float* vg_b2  = (const float*)d_in[12];
  float* out = (float*)d_out;

  float* sums   = (float*)d_ws;      // N*3 floats
  float* counts = sums + 3 * NND;    // N floats

  // Zero scatter accumulators (graph-capture-safe).
  hipMemsetAsync(d_ws, 0, (size_t)4 * NND * sizeof(float), stream);

  const size_t smemEdge = (size_t)W1PITCH * HDIM * 2 + HDIM * 4 + HDIM * 4;
  const size_t smemNode = (size_t)W1PITCH * HDIM * 2 + HDIM * 4 + HDIM * 5 * 4;

  // Allow >default dynamic LDS (no-op/ignored if unnecessary on AMD).
  hipFuncSetAttribute((const void*)edge_weight_scatter,
                      hipFuncAttributeMaxDynamicSharedMemorySize, (int)smemEdge);
  hipFuncSetAttribute((const void*)node_gate_combine,
                      hipFuncAttributeMaxDynamicSharedMemorySize, (int)smemNode);

  edge_weight_scatter<<<1250, 128, smemEdge, stream>>>(
      m_ij, x, eidx, ew_W1, ew_b1, ew_W2, ew_b2, sums, counts);

  node_gate_combine<<<196, 128, smemNode, stream>>>(
      h_in, vel, vg_W1, vg_b1, vg_W2, vg_b2, out);

  finalize_add<<<(3 * NND + 255) / 256, 256, 0, stream>>>(out, sums, counts);

  (void)in_sizes; (void)n_in; (void)out_size; (void)ws_size;
}
